// SelfAttention_28226525069881
// MI455X (gfx1250) — compile-verified, matrix operand order checked by reading, not run on hardware
//
#include <hip/hip_runtime.h>
#include <hip/hip_bf16.h>

// ---------------------------------------------------------------------------
// MI455X (gfx1250) self-attention + LoRA, bf16 WMMA pipeline.
//   - all matrix math on v_wmma_f32_16x16x32_bf16 (fp32 accumulate)
//   - GEMM tiles fed by the Tensor Data Mover (tensor_load_to_lds,
//     double-buffered, s_wait_tensorcnt) issued by wave 0; LDS destination
//     pointer escaped via asm so the compiler keeps the ds_load consumers
//   - attention: streaming softmax, LDS-staged K/V^T, global prefetch
// ---------------------------------------------------------------------------

typedef __attribute__((ext_vector_type(16))) __bf16 v16bf;
typedef __attribute__((ext_vector_type(8)))  __bf16 v8bf;
typedef __attribute__((ext_vector_type(8)))  float  v8f;
typedef __attribute__((ext_vector_type(4)))  unsigned int u32x4;
typedef __attribute__((ext_vector_type(8)))  int i32x8;
typedef __attribute__((ext_vector_type(4)))  int i32x4;

union Frag16 { v16bf v; v8bf h8[2]; };
union BF16x8 { uint4 u; __bf16 h[8]; };

#define D_MODEL 1024
#define HEADS   16
#define DK      64
#define RANK    5
#define BATCH   2
#define SEQ     2048
#define ROWS    (BATCH * SEQ)          // 4096

// ---------------------------------------------------------------------------
// TDM: 2-D tile load (tile_k x tile_rows, bf16) into LDS, per ISA D# layout.
//   group0: count=1 | lds_addr | global_addr | type=2
//   group1: data_size=2B | tensor_dim0/1 | tile_dim0/1 | tensor_dim0_stride
// The LDS destination pointer is escaped into an asm memory clobber so the
// compiler treats the DMA as a real writer of the shared array.
// ---------------------------------------------------------------------------
__device__ __forceinline__ void tdm_load_tile_bf16(const __bf16* gptr,
                                                   __bf16* lds_dst,
                                                   unsigned lds_byte_off,
                                                   unsigned tile_k,
                                                   unsigned tile_rows,
                                                   unsigned tensor_k,
                                                   unsigned tensor_rows)
{
    unsigned long long ga = (unsigned long long)(uintptr_t)gptr;
    u32x4 g0;
    g0[0] = 1u;                                            // count=1, user mode
    g0[1] = lds_byte_off;                                  // lds_addr
    g0[2] = (unsigned)(ga & 0xFFFFFFFFu);                  // global_addr[31:0]
    g0[3] = (unsigned)((ga >> 32) & 0x1FFFFFFu) | (2u << 30);  // addr[56:32] | type=2

    i32x8 g1;
    g1[0] = (int)(1u << 16);                               // wg_mask=0, data_size=2B
    g1[1] = (int)((tensor_k & 0xFFFFu) << 16);             // tensor_dim0[15:0] @48
    g1[2] = (int)(((tensor_k >> 16) & 0xFFFFu) |
                  ((tensor_rows & 0xFFFFu) << 16));        // dim0 hi | dim1 lo
    g1[3] = (int)(((tensor_rows >> 16) & 0xFFFFu) |
                  ((tile_k & 0xFFFFu) << 16));             // dim1 hi | tile_dim0
    g1[4] = (int)(tile_rows & 0xFFFFu);                    // tile_dim1 (tile_dim2=0)
    g1[5] = (int)tensor_k;                                 // dim0_stride[31:0]
    g1[6] = 0;                                             // stride hi | dim1_stride lo
    g1[7] = 0;

    i32x4 z4 = {0, 0, 0, 0};
#if defined(__clang_major__) && (__clang_major__ >= 23)
    i32x8 z8 = {0, 0, 0, 0, 0, 0, 0, 0};
    __builtin_amdgcn_tensor_load_to_lds(g0, g1, z4, z4, z8, 0);
#else
    __builtin_amdgcn_tensor_load_to_lds(g0, g1, z4, z4, 0);
#endif
    // escape: the DMA writes lds_dst behind the compiler's back
    asm volatile("" :: "r"(lds_dst) : "memory");
}

// ---------------------------------------------------------------------------
// Kernel 1: merged LoRA weight  Wm = W + dB @ dW  -> bf16
// ---------------------------------------------------------------------------
__global__ void merge_weights_bf16(const float* __restrict__ W,
                                   const float* __restrict__ dW,
                                   const float* __restrict__ dB,
                                   __bf16* __restrict__ dst)
{
    int idx = blockIdx.x * blockDim.x + threadIdx.x;   // [0, 1024*1024)
    int o = idx >> 10, i = idx & 1023;
    float acc = W[idx];
#pragma unroll
    for (int r = 0; r < RANK; ++r)
        acc += dB[o * RANK + r] * dW[r * D_MODEL + i];
    dst[idx] = (__bf16)acc;
}

// ---------------------------------------------------------------------------
// Kernel 2: fp32 -> bf16 cast
// ---------------------------------------------------------------------------
__global__ void cast_f32_bf16(const float* __restrict__ src,
                              __bf16* __restrict__ dst, int n)
{
    int i = blockIdx.x * blockDim.x + threadIdx.x;
    if (i < n) dst[i] = (__bf16)src[i];
}

// ---------------------------------------------------------------------------
// Kernel 3/5: C[M,N] = A[M,1024] @ W[N,1024]^T + bias
//   128x128 tile / workgroup, 8 wave32 waves, wave = 32x64 (2x4 WMMA frags).
//   K-slabs (128x32 bf16) DMA'd into double-buffered LDS by the Tensor Data
//   Mover (wave 0 issues, s_wait_tensorcnt, workgroup barrier releases).
// ---------------------------------------------------------------------------
__launch_bounds__(256)
__global__ void gemm_bf16_wmma(const __bf16* __restrict__ A,
                               const __bf16* __restrict__ W,
                               int a_rows, int w_rows,
                               const float* __restrict__ bias0,
                               const float* __restrict__ bias1,
                               const float* __restrict__ bias2,
                               __bf16* __restrict__ outb,   // bf16 out (or null)
                               float*  __restrict__ outf)   // f32  out (or null)
{
    // [buf][ A:128x32 | B:128x32 ] ; per-buffer pair = 16 KB, 2 buffers
    __shared__ __align__(16) __bf16 smem[2 * 2 * 128 * 32];

    const int tid  = threadIdx.x;
    const int lane = tid & 31, wid = tid >> 5;
    const int half = lane >> 4, l15 = lane & 15;
    const int wave_m = wid >> 1;                 // 0..3 : 32-row block
    const int wave_n = wid & 1;                  // 0..1 : 64-col block
    const int row0 = blockIdx.y * 128;
    const int col0 = blockIdx.x * 128;
    const int N    = gridDim.x * 128;

    v8f acc[2][4];
#pragma unroll
    for (int mf = 0; mf < 2; ++mf)
#pragma unroll
        for (int nf = 0; nf < 4; ++nf) acc[mf][nf] = v8f{};

    // prologue: DMA first K-slab pair into buffer 0
    if (wid == 0) {
        tdm_load_tile_bf16(A + (size_t)row0 * D_MODEL, smem, 0u,
                           32, 128, D_MODEL, a_rows);
        tdm_load_tile_bf16(W + (size_t)col0 * D_MODEL, smem + 4096, 8192u,
                           32, 128, D_MODEL, w_rows);
    }

    for (int kb = 0; kb < D_MODEL; kb += 32) {
        const int buf = (kb >> 5) & 1;
        if (wid == 0) {
            if (kb + 32 < D_MODEL) {       // DMA next pair into the other buffer
                const int nbuf = buf ^ 1;
                unsigned boff = nbuf ? 16384u : 0u;
                tdm_load_tile_bf16(A + (size_t)row0 * D_MODEL + kb + 32,
                                   smem + nbuf * 8192, boff,
                                   32, 128, D_MODEL, a_rows);
                tdm_load_tile_bf16(W + (size_t)col0 * D_MODEL + kb + 32,
                                   smem + nbuf * 8192 + 4096, boff + 8192u,
                                   32, 128, D_MODEL, w_rows);
                __builtin_amdgcn_s_wait_tensorcnt(2);   // current pair landed
            } else {
                __builtin_amdgcn_s_wait_tensorcnt(0);
            }
        }
        __syncthreads();                   // release compute waves

        const __bf16* sA = smem + buf * 8192;          // elements
        const __bf16* sB = smem + buf * 8192 + 4096;

        Frag16 af[2], bfg[4];
#pragma unroll
        for (int mf = 0; mf < 2; ++mf) {
            int r = wave_m * 32 + mf * 16 + l15;
            af[mf].h8[0] = *(const v8bf*)&sA[r * 32 + half * 8];
            af[mf].h8[1] = *(const v8bf*)&sA[r * 32 + half * 8 + 16];
        }
#pragma unroll
        for (int nf = 0; nf < 4; ++nf) {
            int c = wave_n * 64 + nf * 16 + l15;
            bfg[nf].h8[0] = *(const v8bf*)&sB[c * 32 + half * 8];
            bfg[nf].h8[1] = *(const v8bf*)&sB[c * 32 + half * 8 + 16];
        }
#pragma unroll
        for (int mf = 0; mf < 2; ++mf)
#pragma unroll
            for (int nf = 0; nf < 4; ++nf)
                acc[mf][nf] = __builtin_amdgcn_wmma_f32_16x16x32_bf16(
                    false, af[mf].v, false, bfg[nf].v,
                    (short)0, acc[mf][nf], false, false);
        __syncthreads();                   // all reads done before buffer reuse
    }

    // epilogue: +bias, write bf16 or f32 (C layout: VGPR i -> row i+8*half)
#pragma unroll
    for (int mf = 0; mf < 2; ++mf)
#pragma unroll
        for (int nf = 0; nf < 4; ++nf)
#pragma unroll
            for (int i = 0; i < 8; ++i) {
                int r = row0 + wave_m * 32 + mf * 16 + i + half * 8;
                int c = col0 + wave_n * 64 + nf * 16 + l15;
                int seg = c >> 10, cc = c & 1023;
                float b = (seg == 0) ? bias0[cc] : (seg == 1 ? bias1[cc] : bias2[cc]);
                float v = acc[mf][nf][i] + b;
                if (outb) outb[(size_t)r * N + c] = (__bf16)v;
                else      outf[(size_t)r * N + c] = v;
            }
}

// ---------------------------------------------------------------------------
// Kernel 4: streaming-softmax attention, one (b, h, 128-q-row) per workgroup.
// ---------------------------------------------------------------------------
__launch_bounds__(256)
__global__ void flash_attn_wmma(const __bf16* __restrict__ qkv,   // [4096, 3072]
                                __bf16* __restrict__ attn)        // [4096, 1024]
{
    __shared__ __align__(16) __bf16 sK [32 * 64];
    __shared__ __align__(16) __bf16 sVt[64 * 32];
    __shared__ __align__(16) __bf16 sP [8][16 * 32];

    const int tid  = threadIdx.x, lane = tid & 31, wid = tid >> 5;
    const int half = lane >> 4, l15 = lane & 15;
    const int qt = blockIdx.x;        // q tile (16 of them)
    const int h  = blockIdx.y;        // head
    const int b  = blockIdx.z;        // batch
    const int q0 = qt * 128 + wid * 16;
    const size_t rowb = (size_t)b * SEQ;

    // Q fragments (A layout), loaded once: d = 64 -> two 16x32 frags
    Frag16 qa[2];
    {
        const __bf16* qrow = qkv + (rowb + q0 + l15) * 3072 + h * DK;
#pragma unroll
        for (int d = 0; d < 2; ++d) {
            qa[d].h8[0] = *(const v8bf*)(qrow + d * 32 + half * 8);
            qa[d].h8[1] = *(const v8bf*)(qrow + d * 32 + half * 8 + 16);
        }
    }

    v8f acc[4];
#pragma unroll
    for (int f = 0; f < 4; ++f) acc[f] = v8f{};
    float mrow[8], lrow[8];
#pragma unroll
    for (int i = 0; i < 8; ++i) { mrow[i] = -1e30f; lrow[i] = 0.f; }

    const int skey = tid >> 3;              // 0..31
    const int sd   = (tid & 7) << 3;        // 0,8,..,56

    for (int kc = 0; kc < SEQ; kc += 32) {
        // --- stage K (row-major) and V (transposed) chunk ---
        const __bf16* krow = qkv + (rowb + kc + skey) * 3072 + 1024 + h * DK + sd;
        const __bf16* vrow = qkv + (rowb + kc + skey) * 3072 + 2048 + h * DK + sd;
        uint4 kd = *(const uint4*)krow;
        BF16x8 vd; vd.u = *(const uint4*)vrow;
        if (kc + 32 < SEQ) {                 // prefetch next chunk
            __builtin_prefetch((const void*)(krow + 32 * 3072), 0, 3);
            __builtin_prefetch((const void*)(vrow + 32 * 3072), 0, 3);
        }
        *(uint4*)&sK[skey * 64 + sd] = kd;
#pragma unroll
        for (int j = 0; j < 8; ++j) sVt[(sd + j) * 32 + skey] = vd.h[j];
        __syncthreads();

        // --- scores S = Q @ K^T (16 q x 32 keys) ---
        v8f s0 = v8f{}, s1 = v8f{};
#pragma unroll
        for (int d = 0; d < 2; ++d) {
            Frag16 kb0, kb1;
            kb0.h8[0] = *(const v8bf*)&sK[l15 * 64 + d * 32 + half * 8];
            kb0.h8[1] = *(const v8bf*)&sK[l15 * 64 + d * 32 + half * 8 + 16];
            kb1.h8[0] = *(const v8bf*)&sK[(16 + l15) * 64 + d * 32 + half * 8];
            kb1.h8[1] = *(const v8bf*)&sK[(16 + l15) * 64 + d * 32 + half * 8 + 16];
            s0 = __builtin_amdgcn_wmma_f32_16x16x32_bf16(false, qa[d].v, false, kb0.v,
                                                         (short)0, s0, false, false);
            s1 = __builtin_amdgcn_wmma_f32_16x16x32_bf16(false, qa[d].v, false, kb1.v,
                                                         (short)0, s1, false, false);
        }

        // --- streaming softmax (row = i + 8*half, 16-lane group reductions) ---
        float corr[8];
#pragma unroll
        for (int i = 0; i < 8; ++i) {
            float a0 = s0[i] * 0.125f;          // 1/sqrt(64)
            float a1 = s1[i] * 0.125f;
            float cm = fmaxf(a0, a1);
#pragma unroll
            for (int mlt = 8; mlt >= 1; mlt >>= 1)
                cm = fmaxf(cm, __shfl_xor(cm, mlt, 32));
            float mn = fmaxf(mrow[i], cm);
            float cr = __expf(mrow[i] - mn);
            float p0 = __expf(a0 - mn);
            float p1 = __expf(a1 - mn);
            float rs = p0 + p1;
#pragma unroll
            for (int mlt = 8; mlt >= 1; mlt >>= 1)
                rs += __shfl_xor(rs, mlt, 32);
            lrow[i] = lrow[i] * cr + rs;
            mrow[i] = mn;
            corr[i] = cr;
            // C-layout -> LDS (per-wave) for A-layout reload
            sP[wid][(i + half * 8) * 32 + l15]      = (__bf16)p0;
            sP[wid][(i + half * 8) * 32 + l15 + 16] = (__bf16)p1;
        }
#pragma unroll
        for (int f = 0; f < 4; ++f)
#pragma unroll
            for (int i = 0; i < 8; ++i) acc[f][i] *= corr[i];

        // --- O += P @ V  (contraction over 32 keys) ---
        Frag16 pa;
        pa.h8[0] = *(const v8bf*)&sP[wid][l15 * 32 + half * 8];
        pa.h8[1] = *(const v8bf*)&sP[wid][l15 * 32 + half * 8 + 16];
#pragma unroll
        for (int f = 0; f < 4; ++f) {
            Frag16 vb;
            int c = f * 16 + l15;
            vb.h8[0] = *(const v8bf*)&sVt[c * 32 + half * 8];
            vb.h8[1] = *(const v8bf*)&sVt[c * 32 + half * 8 + 16];
            acc[f] = __builtin_amdgcn_wmma_f32_16x16x32_bf16(false, pa.v, false, vb.v,
                                                             (short)0, acc[f], false, false);
        }
        __syncthreads();
    }

    // normalize and write bf16 attention output [row, h*64 + d]
#pragma unroll
    for (int f = 0; f < 4; ++f)
#pragma unroll
        for (int i = 0; i < 8; ++i) {
            int r = q0 + i + half * 8;
            int c = h * DK + f * 16 + l15;
            attn[(rowb + r) * D_MODEL + c] = (__bf16)(acc[f][i] / lrow[i]);
        }
}

// ---------------------------------------------------------------------------
// Launch
// ---------------------------------------------------------------------------
extern "C" void kernel_launch(void* const* d_in, const int* in_sizes, int n_in,
                              void* d_out, int out_size, void* d_ws, size_t ws_size,
                              hipStream_t stream)
{
    const float* x    = (const float*)d_in[0];
    const float* q_w  = (const float*)d_in[1];
    const float* q_b  = (const float*)d_in[2];
    const float* q_dw = (const float*)d_in[3];
    const float* q_db = (const float*)d_in[4];
    const float* k_w  = (const float*)d_in[5];
    const float* k_b  = (const float*)d_in[6];
    const float* k_dw = (const float*)d_in[7];
    const float* k_db = (const float*)d_in[8];
    const float* v_w  = (const float*)d_in[9];
    const float* v_b  = (const float*)d_in[10];
    const float* v_dw = (const float*)d_in[11];
    const float* v_db = (const float*)d_in[12];
    const float* o_w  = (const float*)d_in[13];
    const float* o_b  = (const float*)d_in[14];
    const float* o_dw = (const float*)d_in[15];
    const float* o_db = (const float*)d_in[16];

    // workspace layout (bytes)
    char* ws = (char*)d_ws;
    __bf16* Wqkv = (__bf16*)(ws);                        //  [3072,1024]  6 MB
    __bf16* Wo   = (__bf16*)(ws + 6291456);              //  [1024,1024]  2 MB
    __bf16* Xb   = (__bf16*)(ws + 8388608);              //  [4096,1024]  8 MB
    __bf16* QKV  = (__bf16*)(ws + 16777216);             //  [4096,3072] 24 MB
    __bf16* AT   = (__bf16*)(ws + 41943040);             //  [4096,1024]  8 MB

    dim3 blk(256);
    merge_weights_bf16<<<4096, blk, 0, stream>>>(q_w, q_dw, q_db, Wqkv);
    merge_weights_bf16<<<4096, blk, 0, stream>>>(k_w, k_dw, k_db, Wqkv + 1024 * 1024);
    merge_weights_bf16<<<4096, blk, 0, stream>>>(v_w, v_dw, v_db, Wqkv + 2 * 1024 * 1024);
    merge_weights_bf16<<<4096, blk, 0, stream>>>(o_w, o_dw, o_db, Wo);

    cast_f32_bf16<<<16384, blk, 0, stream>>>(x, Xb, ROWS * D_MODEL);

    // QKV = Xb @ Wqkv^T + [qb|kb|vb]   -> bf16 [4096, 3072]
    gemm_bf16_wmma<<<dim3(24, 32), blk, 0, stream>>>(Xb, Wqkv, ROWS, 3072,
                                                     q_b, k_b, v_b, QKV, nullptr);

    // attention -> bf16 [4096, 1024]
    flash_attn_wmma<<<dim3(16, 16, 2), blk, 0, stream>>>(QKV, AT);

    // out = AT @ Wo^T + ob  -> f32 d_out
    gemm_bf16_wmma<<<dim3(8, 32), blk, 0, stream>>>(AT, Wo, ROWS, 1024,
                                                    o_b, o_b, o_b,
                                                    nullptr, (float*)d_out);
}
